// CausalSelfAttention_1168231104549
// MI455X (gfx1250) — matrix operand order repaired and checked by reading, not compiled, not run on hardware
//
#include <hip/hip_runtime.h>

// ---------------- problem constants ----------------
#define Bn   4
#define Tn   2048
#define Cn   1024
#define Hn   16
#define Dn   64
#define NQKV 3072
#define Mrows (Bn * Tn)   // 8192

typedef __attribute__((ext_vector_type(16))) __bf16 v16bf;
typedef __attribute__((ext_vector_type(8)))  float  v8f;

union ABf {
    v16bf v;
    uint4 q[2];
    unsigned short u[16];
};

__device__ __forceinline__ unsigned short f32_to_bf16_bits(float f) {
    union { float f; unsigned int u; } c;
    c.f = f;
    unsigned int u = c.u;
    u += 0x7FFFu + ((u >> 16) & 1u);   // round to nearest even
    return (unsigned short)(u >> 16);
}

__device__ __forceinline__ unsigned int pack2_bf16(float lo, float hi) {
    return (((unsigned int)f32_to_bf16_bits(hi)) << 16) | (unsigned int)f32_to_bf16_bits(lo);
}

__device__ __forceinline__ v8f wmma_bf16(const v16bf& a, const v16bf& b, const v8f& c) {
    return __builtin_amdgcn_wmma_f32_16x16x32_bf16(
        /*neg_a=*/false, a, /*neg_b=*/false, b,
        /*c_mod=*/(short)0, c, /*reuse_a=*/false, /*reuse_b=*/false);
}

// =====================================================================
// Shared GEMM block mainloop: block tile 128(M) x 256(N), 8 waves in a
// 2x4 grid, each wave 64x64 via 4x4 WMMA accumulators (16 WMMA per
// k-step per wave for 16 b128 LDS loads). A,B staged to LDS as bf16;
// B staged transposed (k-contiguous) so every fragment is 2x b128.
// =====================================================================
__device__ __forceinline__ void gemm_block_128x256(const float* __restrict__ A, int lda,
                                                   const float* __restrict__ Bm, int ldb,
                                                   int rowBlk, int colBlk, int K,
                                                   v8f acc[4][4]) {
    __shared__ __align__(16) unsigned short Ash[128][40];   // [m][k], 80B row stride
    __shared__ __align__(16) unsigned short Bsh[256][40];   // [n][k] (transposed)

    const int tid  = threadIdx.x;
    const int lane = tid & 31;
    const int wid  = tid >> 5;
    const int n16  = lane & 15;
    const int g    = lane >> 4;
    const int rw   = (wid >> 2) * 64;   // wave M offset in block tile
    const int cw   = (wid & 3) * 64;    // wave N offset in block tile

    // staging assignments
    const int arow = tid & 127;         // A: row, 16 k-values each
    const int ah   = tid >> 7;          // 0..1 -> k offset 16*ah
    const int bn   = tid;               // B: one column, all 32 k-values

    for (int k0 = 0; k0 < K; k0 += 32) {
        // ---- stage A tile (128x32 f32 -> bf16), vectorized ----
        {
            const float4* ap = (const float4*)(A + (size_t)(rowBlk + arow) * lda + k0 + ah * 16);
            const float4 v0 = ap[0], v1 = ap[1], v2 = ap[2], v3 = ap[3];
            uint4 p0, p1;
            p0.x = pack2_bf16(v0.x, v0.y);
            p0.y = pack2_bf16(v0.z, v0.w);
            p0.z = pack2_bf16(v1.x, v1.y);
            p0.w = pack2_bf16(v1.z, v1.w);
            p1.x = pack2_bf16(v2.x, v2.y);
            p1.y = pack2_bf16(v2.z, v2.w);
            p1.z = pack2_bf16(v3.x, v3.y);
            p1.w = pack2_bf16(v3.z, v3.w);
            *(uint4*)&Ash[arow][ah * 16]     = p0;
            *(uint4*)&Ash[arow][ah * 16 + 8] = p1;
        }
        // ---- stage B tile (32x256 f32 -> bf16, transposed into [n][k]) ----
        {
            const float* bp = Bm + (size_t)k0 * ldb + (colBlk + bn);
            float v[32];
#pragma unroll
            for (int i = 0; i < 32; ++i) v[i] = bp[(size_t)i * ldb];
            uint4 pk[4];
#pragma unroll
            for (int qq = 0; qq < 4; ++qq) {
                pk[qq].x = pack2_bf16(v[qq * 8 + 0], v[qq * 8 + 1]);
                pk[qq].y = pack2_bf16(v[qq * 8 + 2], v[qq * 8 + 3]);
                pk[qq].z = pack2_bf16(v[qq * 8 + 4], v[qq * 8 + 5]);
                pk[qq].w = pack2_bf16(v[qq * 8 + 6], v[qq * 8 + 7]);
            }
#pragma unroll
            for (int qq = 0; qq < 4; ++qq)
                *(uint4*)&Bsh[bn][qq * 8] = pk[qq];
        }
        __syncthreads();

        // ---- fragments: two b128 LDS loads each ----
        ABf a[4], b[4];
#pragma unroll
        for (int mi = 0; mi < 4; ++mi) {
            const unsigned short* ar = &Ash[rw + mi * 16 + n16][0];
            a[mi].q[0] = *(const uint4*)(ar + 16 * g);       // K = 16g .. 16g+7
            a[mi].q[1] = *(const uint4*)(ar + 16 * g + 8);   // K = 16g+8 .. +15
        }
#pragma unroll
        for (int ni = 0; ni < 4; ++ni) {
            const unsigned short* br = &Bsh[cw + ni * 16 + n16][0];
            b[ni].q[0] = *(const uint4*)(br + 16 * g);       // K = 16g .. 16g+7
            b[ni].q[1] = *(const uint4*)(br + 16 * g + 8);   // K = 16g+8 .. +15
        }
#pragma unroll
        for (int mi = 0; mi < 4; ++mi)
#pragma unroll
            for (int ni = 0; ni < 4; ++ni)
                acc[mi][ni] = wmma_bf16(a[mi].v, b[ni].v, acc[mi][ni]);
        __syncthreads();
    }
}

// =====================================================================
// Kernel 1: qkv = x @ Wqkv + bqkv.
// Q,K scattered to bf16 [B,H,T,D] (d-contiguous); V to bf16 [B,H,D,T]
// (t-contiguous). 1/sqrt(D) folded into Q.
// =====================================================================
__global__ void __launch_bounds__(256)
qkv_gemm_kernel(const float* __restrict__ x, const float* __restrict__ Wqkv,
                const float* __restrict__ bqkv,
                unsigned short* __restrict__ Qbf, unsigned short* __restrict__ Kbf,
                unsigned short* __restrict__ Vt) {
    const int tid  = threadIdx.x;
    const int lane = tid & 31;
    const int wid  = tid >> 5;
    const int n16  = lane & 15;
    const int g    = lane >> 4;
    const int row0 = blockIdx.y * 128 + (wid >> 2) * 64;
    const int col0 = blockIdx.x * 256 + (wid & 3) * 64;

    v8f acc[4][4] = {};
    gemm_block_128x256(x, Cn, Wqkv, NQKV, blockIdx.y * 128, blockIdx.x * 256, Cn, acc);

#pragma unroll
    for (int mi = 0; mi < 4; ++mi) {
#pragma unroll
        for (int ni = 0; ni < 4; ++ni) {
            const int col  = col0 + ni * 16 + n16;
            const int sect = col >> 10;            // 0:q 1:k 2:v (uniform per tile)
            const int cc   = col & (Cn - 1);
            const int h    = cc >> 6;
            const int d    = cc & (Dn - 1);
            const float bias = bqkv[col];
            if (sect == 2) {
                // V: pack 8 consecutive t rows -> one b128 store into [B,H,D,T]
                const int trow0 = row0 + mi * 16 + 8 * g;
                const int bidx  = trow0 >> 11;
                const int t0    = trow0 & (Tn - 1);
                float vv[8];
#pragma unroll
                for (int r = 0; r < 8; ++r) vv[r] = acc[mi][ni][r] + bias;
                uint4 pk;
                pk.x = pack2_bf16(vv[0], vv[1]);
                pk.y = pack2_bf16(vv[2], vv[3]);
                pk.z = pack2_bf16(vv[4], vv[5]);
                pk.w = pack2_bf16(vv[6], vv[7]);
                *(uint4*)&Vt[((size_t)(bidx * Hn + h) * Dn + d) * Tn + t0] = pk;
            } else {
                unsigned short* dst = (sect == 0) ? Qbf : Kbf;
                const float qscale = (sect == 0) ? 0.125f : 1.0f;
#pragma unroll
                for (int r = 0; r < 8; ++r) {
                    const int trow = row0 + mi * 16 + r + 8 * g;
                    const int bidx = trow >> 11;
                    const int t    = trow & (Tn - 1);
                    const float v  = (acc[mi][ni][r] + bias) * qscale;
                    dst[((size_t)(bidx * Hn + h) * Tn + t) * Dn + d] = f32_to_bf16_bits(v);
                }
            }
        }
    }
}

// =====================================================================
// Kernel 2: causal flash attention. One wave per 32-query strip (two
// 16-row q-tiles sharing K/V fragments), 64-key chunks:
// 16 WMMAs for S + 16 WMMAs for P*V per chunk. All fragments 2x b128.
// =====================================================================
__global__ void __launch_bounds__(256)
attn_kernel(const unsigned short* __restrict__ Qbf, const unsigned short* __restrict__ Kbf,
            const unsigned short* __restrict__ Vt, float* __restrict__ Obuf) {
    __shared__ __align__(16) unsigned short pbuf[8][32][72];  // per-wave P (bf16), 144B stride

    const int tid  = threadIdx.x;
    const int lane = tid & 31;
    const int wid  = tid >> 5;
    const int n16  = lane & 15;
    const int g    = lane >> 4;
    const int bh   = blockIdx.x;               // 0..63
    const int b    = bh >> 4;
    const int h    = bh & (Hn - 1);
    const int qBase = blockIdx.y * 256 + wid * 32;
    const float NEG_INF = -__builtin_inff();

    const unsigned short* Qp  = Qbf + (size_t)bh * Tn * Dn;   // [t][d]
    const unsigned short* Kp  = Kbf + (size_t)bh * Tn * Dn;   // [t][d]
    const unsigned short* Vtp = Vt  + (size_t)bh * Dn * Tn;   // [d][t]

    // Q fragments: per q-tile, A-layout 16x32, two d-halves; each = 2x b128
    ABf aq[2][2];
#pragma unroll
    for (int qt = 0; qt < 2; ++qt) {
        const unsigned short* qr = Qp + (size_t)(qBase + qt * 16 + n16) * Dn;
#pragma unroll
        for (int f = 0; f < 2; ++f) {
            aq[qt][f].q[0] = *(const uint4*)(qr + 32 * f + 16 * g);
            aq[qt][f].q[1] = *(const uint4*)(qr + 32 * f + 16 * g + 8);
        }
    }

    v8f o[2][4] = {};
    float mrow[2][8], lrow[2][8];
#pragma unroll
    for (int qt = 0; qt < 2; ++qt)
#pragma unroll
        for (int r = 0; r < 8; ++r) { mrow[qt][r] = NEG_INF; lrow[qt][r] = 0.0f; }

    const int nChunks = (qBase + 32 + 63) >> 6;   // causal: keys 0 .. qBase+31
    for (int c = 0; c < nChunks; ++c) {
        const int k0 = c << 6;

        if (c + 1 < nChunks) {   // pull next K/V chunk toward the WGP
            __builtin_prefetch(Kp + (size_t)(k0 + 64) * Dn, 0, 1);
            __builtin_prefetch(Vtp + (k0 + 64), 0, 1);
        }

        // ---- S = Q K^T : K frags shared by both q-tiles ----
        v8f s[2][4] = {};
#pragma unroll
        for (int f = 0; f < 2; ++f) {
            ABf bk[4];
#pragma unroll
            for (int nt = 0; nt < 4; ++nt) {
                const unsigned short* kr =
                    Kp + (size_t)(k0 + nt * 16 + n16) * Dn + 32 * f + 16 * g;
                bk[nt].q[0] = *(const uint4*)(kr);
                bk[nt].q[1] = *(const uint4*)(kr + 8);
            }
#pragma unroll
            for (int qt = 0; qt < 2; ++qt)
#pragma unroll
                for (int nt = 0; nt < 4; ++nt)
                    s[qt][nt] = wmma_bf16(aq[qt][f].v, bk[nt].v, s[qt][nt]);
        }

        // ---- online softmax per q-tile (row m = r + 8g in this lane's half) ----
#pragma unroll
        for (int qt = 0; qt < 2; ++qt) {
#pragma unroll
            for (int r = 0; r < 8; ++r) {
                const int qg = qBase + qt * 16 + r + 8 * g;
                bool  ok[4];
                float cm = NEG_INF;
#pragma unroll
                for (int nt = 0; nt < 4; ++nt) {
                    ok[nt] = (k0 + nt * 16 + n16) <= qg;
                    cm = fmaxf(cm, ok[nt] ? s[qt][nt][r] : NEG_INF);
                }
#pragma unroll
                for (int off = 8; off >= 1; off >>= 1)
                    cm = fmaxf(cm, __shfl_xor(cm, off, 32));
                const float nm    = fmaxf(mrow[qt][r], cm);
                const float alpha = __expf(mrow[qt][r] - nm);
                float ps = 0.0f;
#pragma unroll
                for (int nt = 0; nt < 4; ++nt) {
                    const float p = ok[nt] ? __expf(s[qt][nt][r] - nm) : 0.0f;
                    ps += p;
                    pbuf[wid][qt * 16 + r + 8 * g][nt * 16 + n16] = f32_to_bf16_bits(p);
                }
#pragma unroll
                for (int off = 8; off >= 1; off >>= 1)
                    ps += __shfl_xor(ps, off, 32);
                lrow[qt][r] = lrow[qt][r] * alpha + ps;
                mrow[qt][r] = nm;
#pragma unroll
                for (int dt = 0; dt < 4; ++dt) o[qt][dt][r] *= alpha;
            }
        }

        // wave-local LDS ordering fence (DS ops are in-order per wave)
        asm volatile("s_wait_dscnt 0x0" ::: "memory");

        // ---- O += P V : V frags shared by both q-tiles ----
#pragma unroll
        for (int f = 0; f < 2; ++f) {
            ABf bv[4];
#pragma unroll
            for (int dt = 0; dt < 4; ++dt) {
                const unsigned short* vr =
                    Vtp + (size_t)(dt * 16 + n16) * Tn + k0 + 32 * f + 16 * g;
                bv[dt].q[0] = *(const uint4*)(vr);
                bv[dt].q[1] = *(const uint4*)(vr + 8);
            }
#pragma unroll
            for (int qt = 0; qt < 2; ++qt) {
                ABf ap;
                const unsigned short* pr = &pbuf[wid][qt * 16 + n16][32 * f];
                ap.q[0] = *(const uint4*)(pr + 16 * g);
                ap.q[1] = *(const uint4*)(pr + 16 * g + 8);
#pragma unroll
                for (int dt = 0; dt < 4; ++dt)
                    o[qt][dt] = wmma_bf16(ap.v, bv[dt].v, o[qt][dt]);
            }
        }
    }

    // ---- normalize, write O into [B,T,C] f32 (row=b*T+t, col=h*64+d) ----
#pragma unroll
    for (int qt = 0; qt < 2; ++qt)
#pragma unroll
        for (int r = 0; r < 8; ++r) {
            const int t = qBase + qt * 16 + r + 8 * g;
            const float inv = 1.0f / lrow[qt][r];
            float* orow = Obuf + ((size_t)(b * Tn + t)) * Cn + h * Dn;
#pragma unroll
            for (int dt = 0; dt < 4; ++dt)
                orow[dt * 16 + n16] = o[qt][dt][r] * inv;
        }
}

// =====================================================================
// Kernel 3: out = O @ Wo + bo
// =====================================================================
__global__ void __launch_bounds__(256)
out_gemm_kernel(const float* __restrict__ Obuf, const float* __restrict__ Wo,
                const float* __restrict__ bo, float* __restrict__ out) {
    const int tid  = threadIdx.x;
    const int lane = tid & 31;
    const int wid  = tid >> 5;
    const int n16  = lane & 15;
    const int g    = lane >> 4;
    const int row0 = blockIdx.y * 128 + (wid >> 2) * 64;
    const int col0 = blockIdx.x * 256 + (wid & 3) * 64;

    v8f acc[4][4] = {};
    gemm_block_128x256(Obuf, Cn, Wo, Cn, blockIdx.y * 128, blockIdx.x * 256, Cn, acc);

#pragma unroll
    for (int mi = 0; mi < 4; ++mi)
#pragma unroll
        for (int ni = 0; ni < 4; ++ni) {
            const int col = col0 + ni * 16 + n16;
            const float bias = bo[col];
#pragma unroll
            for (int r = 0; r < 8; ++r) {
                const int trow = row0 + mi * 16 + r + 8 * g;
                out[(size_t)trow * Cn + col] = acc[mi][ni][r] + bias;
            }
        }
}

// ---------------- launch ----------------
extern "C" void kernel_launch(void* const* d_in, const int* in_sizes, int n_in,
                              void* d_out, int out_size, void* d_ws, size_t ws_size,
                              hipStream_t stream) {
    const float* x    = (const float*)d_in[0];
    const float* Wqkv = (const float*)d_in[1];
    const float* bqkv = (const float*)d_in[2];
    const float* Wo   = (const float*)d_in[3];
    const float* bo   = (const float*)d_in[4];
    float* out = (float*)d_out;

    const size_t BHTD = (size_t)Bn * Hn * Tn * Dn;     // 8M elements
    unsigned short* Qbf = (unsigned short*)d_ws;        // 16 MB, [B,H,T,D]
    unsigned short* Kbf = Qbf + BHTD;                   // 16 MB, [B,H,T,D]
    unsigned short* Vt  = Kbf + BHTD;                   // 16 MB, [B,H,D,T]
    float* Obuf = (float*)(Vt + BHTD);                  // 32 MB, [B*T, C]

    qkv_gemm_kernel<<<dim3(NQKV / 256, Mrows / 128), 256, 0, stream>>>(x, Wqkv, bqkv, Qbf, Kbf, Vt);
    attn_kernel<<<dim3(Bn * Hn, Tn / 256), 256, 0, stream>>>(Qbf, Kbf, Vt, Obuf);
    out_gemm_kernel<<<dim3(Cn / 256, Mrows / 128), 256, 0, stream>>>(Obuf, Wo, bo, out);
}
